// OutputLayer_18124761989377
// MI455X (gfx1250) — matrix operand order repaired and checked by reading, not compiled, not run on hardware
//
#include <hip/hip_runtime.h>
#include <hip/hip_bf16.h>
#include <math.h>

#define B_ 8
#define L_ 2500
#define H_ 512
#define Y_ 8921

#define WAVES 8
#define ROWS_PER_WG (WAVES * 16)          // 128 Y rows per workgroup
#define RS 520                            // padded LDS row stride (elements) -> bank spread
#define LC 16                             // L chunk (one WMMA N tile)
#define NCHUNK ((L_ + LC - 1) / LC)       // 157
#define KSTEPS (H_ / 32)                  // 16

typedef __attribute__((ext_vector_type(16))) __bf16 v16bf;
typedef __attribute__((ext_vector_type(8)))  __bf16 bf16x8;
typedef __attribute__((ext_vector_type(8)))  float  v8f;

union ABOp { v16bf v; bf16x8 h[2]; };

__device__ __forceinline__ bf16x8 cvt8g(const float* src) {
  const float4* s4 = (const float4*)src;
  float4 a = s4[0], b = s4[1];
  bf16x8 o;
  o[0] = (__bf16)a.x; o[1] = (__bf16)a.y; o[2] = (__bf16)a.z; o[3] = (__bf16)a.w;
  o[4] = (__bf16)b.x; o[5] = (__bf16)b.y; o[6] = (__bf16)b.z; o[7] = (__bf16)b.w;
  return o;
}

__device__ __forceinline__ void cvt_store8(__bf16* dst, const float* src) {
  *(bf16x8*)dst = cvt8g(src);
}

__device__ __forceinline__ void cvt_store8v(__bf16* dst, float4 a, float4 b) {
  bf16x8 o;
  o[0] = (__bf16)a.x; o[1] = (__bf16)a.y; o[2] = (__bf16)a.z; o[3] = (__bf16)a.w;
  o[4] = (__bf16)b.x; o[5] = (__bf16)b.y; o[6] = (__bf16)b.z; o[7] = (__bf16)b.w;
  *(bf16x8*)dst = o;
}

extern "C" __global__ __launch_bounds__(256)
void attn_pool_kernel(const float* __restrict__ x,
                      const float* __restrict__ U,
                      const float* __restrict__ FW,
                      const float* __restrict__ bias,
                      float* __restrict__ out) {
  extern __shared__ __align__(16) char smem[];
  __bf16* Xl0 = (__bf16*)smem;                    // [LC][RS]  ping
  __bf16* Xl1 = Xl0 + LC * RS;                    // [LC][RS]  pong

  const int b    = blockIdx.y;
  const int y0   = blockIdx.x * ROWS_PER_WG;
  const int tid  = threadIdx.x;
  const int wave = tid >> 5;
  const int lane = tid & 31;
  const int idx  = lane & 15;                     // M row / N col within tile
  const int half = lane >> 4;                     // K-half selector

  const int xr16 = tid >> 4;                      // staging: row within chunk
  const int xc0  = (tid & 15) * 32;               // staging: 32 consecutive H

  // ---- weight-stationary A operands: whole K of U and FW in registers ----
  // A layout: lanes 0-15 row=idx K{k0..k0+7, k0+16..k0+23};
  //           lanes 16-31 same row, K shifted by +8 (half*8)
  ABOp au[KSTEPS], af[KSTEPS];
  {
    const int gy = y0 + wave * 16 + idx;
    if (gy < Y_) {
      const float* Ug = U  + (size_t)gy * H_ + half * 8;
      const float* Fg = FW + (size_t)gy * H_ + half * 8;
      #pragma unroll
      for (int kk = 0; kk < KSTEPS; ++kk) {
        au[kk].h[0] = cvt8g(Ug + kk * 32);
        au[kk].h[1] = cvt8g(Ug + kk * 32 + 16);
        af[kk].h[0] = cvt8g(Fg + kk * 32);
        af[kk].h[1] = cvt8g(Fg + kk * 32 + 16);
      }
    } else {
      bf16x8 z = {};
      #pragma unroll
      for (int kk = 0; kk < KSTEPS; ++kk) {
        au[kk].h[0] = z; au[kk].h[1] = z;
        af[kk].h[0] = z; af[kk].h[1] = z;
      }
    }
  }

  // ---- stage chunk 0 into ping buffer (l0 = 0, always in range) ----
  {
    __bf16* dx = Xl0 + xr16 * RS + xc0;
    const float* sx = x + ((size_t)b * L_ + xr16) * H_ + xc0;
    #pragma unroll
    for (int j = 0; j < 4; ++j) cvt_store8(dx + j * 8, sx + j * 8);
  }
  __syncthreads();

  // per-lane online stats: this lane tracks rows (r + 8*half) of its wave tile
  float Mst[8], Zst[8], Wst[8];
  #pragma unroll
  for (int r = 0; r < 8; ++r) { Mst[r] = -__builtin_inff(); Zst[r] = 0.f; Wst[r] = 0.f; }

  // B operand: lanes 0-15 col=idx K{0..7,16..23}; lanes 16-31 col=idx K{8..15,24..31}
  const int xoff = idx * RS + half * 8;

  for (int ch = 0; ch < NCHUNK; ++ch) {
    const int cur = ch & 1;
    const __bf16* Xa = (cur ? Xl1 : Xl0) + xoff;

    // --- issue next-chunk global loads early (latency hidden behind WMMAs) ---
    const bool havePref = (ch + 1 < NCHUNK);
    bool prefValid = false;
    float4 pv[8];
    if (havePref) {
      int lrow = (ch + 1) * LC + xr16;
      prefValid = lrow < L_;
      if (prefValid) {
        const float4* sx = (const float4*)(x + ((size_t)b * L_ + lrow) * H_ + xc0);
        #pragma unroll
        for (int j = 0; j < 8; ++j) pv[j] = sx[j];
      }
    }

    // --- WMMA K-loop: only B streams from LDS (prefetch distance 2) ---
    v8f s = {}; v8f t = {};
    ABOp bx[3];
    bx[0].h[0] = *(const bf16x8*)(Xa);
    bx[0].h[1] = *(const bf16x8*)(Xa + 16);
    bx[1].h[0] = *(const bf16x8*)(Xa + 32);
    bx[1].h[1] = *(const bf16x8*)(Xa + 48);
    #pragma unroll
    for (int kk = 0; kk < KSTEPS; ++kk) {
      const int c = kk % 3;
      if (kk + 2 < KSTEPS) {
        const int p  = (kk + 2) % 3;
        const int k2 = (kk + 2) * 32;
        bx[p].h[0] = *(const bf16x8*)(Xa + k2);
        bx[p].h[1] = *(const bf16x8*)(Xa + k2 + 16);
      }
      s = __builtin_amdgcn_wmma_f32_16x16x32_bf16(false, au[kk].v, false, bx[c].v, (short)0, s, false, false);
      t = __builtin_amdgcn_wmma_f32_16x16x32_bf16(false, af[kk].v, false, bx[c].v, (short)0, t, false, false);
    }

    // --- per-lane online softmax-weighted accumulation (no cross-lane traffic) ---
    const bool colValid = (ch * LC + idx) < L_;
    #pragma unroll
    for (int r = 0; r < 8; ++r) {
      float sv    = colValid ? s[r] : -__builtin_inff();
      float mn    = fmaxf(Mst[r], sv);
      float scale = __expf(Mst[r] - mn);          // exp(-inf)=0 handles init
      float p     = __expf(sv - mn);              // masked col -> 0
      Zst[r] = Zst[r] * scale + p;
      Wst[r] = Wst[r] * scale + p * t[r];
      Mst[r] = mn;
    }

    // --- drain prefetched chunk into the other LDS buffer ---
    if (havePref) {
      __bf16* dx = (cur ? Xl0 : Xl1) + xr16 * RS + xc0;
      if (prefValid) {
        #pragma unroll
        for (int j = 0; j < 4; ++j) cvt_store8v(dx + j * 8, pv[2 * j], pv[2 * j + 1]);
      } else {
        bf16x8 z = {};
        #pragma unroll
        for (int j = 0; j < 4; ++j) *(bf16x8*)(dx + j * 8) = z;
      }
    }
    __syncthreads();                              // one barrier per chunk
  }

  // ---- cross-lane log-sum-exp merge within 16-lane groups, then write y ----
  #pragma unroll
  for (int r = 0; r < 8; ++r) {
    float mg = Mst[r];
    mg = fmaxf(mg, __shfl_xor(mg, 1, 16));
    mg = fmaxf(mg, __shfl_xor(mg, 2, 16));
    mg = fmaxf(mg, __shfl_xor(mg, 4, 16));
    mg = fmaxf(mg, __shfl_xor(mg, 8, 16));
    float sc = __expf(Mst[r] - mg);
    float zs = Zst[r] * sc, ws = Wst[r] * sc;
    zs += __shfl_xor(zs, 1, 16); ws += __shfl_xor(ws, 1, 16);
    zs += __shfl_xor(zs, 2, 16); ws += __shfl_xor(ws, 2, 16);
    zs += __shfl_xor(zs, 4, 16); ws += __shfl_xor(ws, 4, 16);
    zs += __shfl_xor(zs, 8, 16); ws += __shfl_xor(ws, 8, 16);
    int gy = y0 + wave * 16 + r + 8 * half;       // C-layout: VGPR r -> rows r / r+8
    if (idx == 0 && gy < Y_) {
      out[(size_t)b * Y_ + gy] = ws / zs + bias[gy];
    }
  }
}

// Deterministic single-block BCE-with-logits reduction (no float atomics).
extern "C" __global__ void bce_loss_kernel(const float* __restrict__ y,
                                           const float* __restrict__ target,
                                           float* __restrict__ loss) {
  __shared__ float red[256];
  float acc = 0.f;
  for (int i = threadIdx.x; i < B_ * Y_; i += 256) {
    float v = y[i], tg = target[i];
    acc += fmaxf(v, 0.f) - v * tg + log1pf(__expf(-fabsf(v)));
  }
  red[threadIdx.x] = acc;
  __syncthreads();
  for (int s = 128; s > 0; s >>= 1) {
    if ((int)threadIdx.x < s) red[threadIdx.x] += red[threadIdx.x + s];
    __syncthreads();
  }
  if (threadIdx.x == 0) loss[0] = red[0] / (float)(B_ * Y_);
}

extern "C" void kernel_launch(void* const* d_in, const int* in_sizes, int n_in,
                              void* d_out, int out_size, void* d_ws, size_t ws_size,
                              hipStream_t stream) {
  const float* x  = (const float*)d_in[0];
  const float* tg = (const float*)d_in[1];
  const float* U  = (const float*)d_in[2];
  const float* FW = (const float*)d_in[3];
  const float* bs = (const float*)d_in[4];
  float* out = (float*)d_out;

  // LDS: 2 x-chunk ping-pong buffers only: 2 * 16 * 520 * 2 = 33,280 B
  const size_t smem = (size_t)(2 * LC * RS) * sizeof(__bf16);
  (void)hipFuncSetAttribute((const void*)attn_pool_kernel,
                            hipFuncAttributeMaxDynamicSharedMemorySize, (int)smem);

  dim3 grid((Y_ + ROWS_PER_WG - 1) / ROWS_PER_WG, B_);   // 70 x 8
  attn_pool_kernel<<<grid, 256, smem, stream>>>(x, U, FW, bs, out);
  bce_loss_kernel<<<1, 256, 0, stream>>>(out, tg, out + (size_t)B_ * Y_);
}
